// ScaledDotProductAttention_77730318123152
// MI455X (gfx1250) — matrix operand order repaired and checked by reading, not compiled, not run on hardware
//
#include <hip/hip_runtime.h>

typedef __attribute__((ext_vector_type(16))) _Float16 v16h;
typedef __attribute__((ext_vector_type(8)))  _Float16 v8h;
typedef __attribute__((ext_vector_type(8)))  float    v8f;
typedef __attribute__((ext_vector_type(4)))  int      v4i;

#define BATCH 16
#define SEQ   2048
#define DIM   64
#define NTILE (SEQ / 16)    // 128 K-tiles of 16 rows
#define NPAIR (NTILE / 2)   // 64 pairs (K=32 per WMMA chain)
#define WAVES 8
#define QROWS_WG (WAVES * 16)  // 128 q rows per workgroup
#define VT_STRIDE 40           // halves; 80B row stride -> 16B aligned fragments
#define PS_STRIDE 40
#define NELEM ((size_t)BATCH * SEQ * DIM)   // 2,097,152

#if defined(__has_builtin)
#if __has_builtin(__builtin_amdgcn_global_load_async_to_lds_b128)
#define HAVE_ASYNC_LDS 1
#endif
#endif
#ifndef HAVE_ASYNC_LDS
#define HAVE_ASYNC_LDS 0
#endif

typedef v4i __attribute__((address_space(1)))* gptr_v4i;
typedef v4i __attribute__((address_space(3)))* lptr_v4i;

__device__ __forceinline__ void wait_async_lds() {
#if HAVE_ASYNC_LDS
#if defined(__has_builtin) && __has_builtin(__builtin_amdgcn_s_wait_asynccnt)
  __builtin_amdgcn_s_wait_asynccnt(0);
#else
  asm volatile("s_wait_asynccnt 0x0" ::: "memory");
#endif
#endif
}

// ---------- helpers ----------

// 16 contiguous f32 -> v16h (for the f32 fallback path)
__device__ __forceinline__ v16h cvt16(const float* p) {
  float4 f0 = ((const float4*)p)[0];
  float4 f1 = ((const float4*)p)[1];
  float4 f2 = ((const float4*)p)[2];
  float4 f3 = ((const float4*)p)[3];
  v16h r;
  r[0]  = (_Float16)f0.x; r[1]  = (_Float16)f0.y; r[2]  = (_Float16)f0.z; r[3]  = (_Float16)f0.w;
  r[4]  = (_Float16)f1.x; r[5]  = (_Float16)f1.y; r[6]  = (_Float16)f1.z; r[7]  = (_Float16)f1.w;
  r[8]  = (_Float16)f2.x; r[9]  = (_Float16)f2.y; r[10] = (_Float16)f2.z; r[11] = (_Float16)f2.w;
  r[12] = (_Float16)f3.x; r[13] = (_Float16)f3.y; r[14] = (_Float16)f3.z; r[15] = (_Float16)f3.w;
  return r;
}

// A-fragment (16x32 f16): lane row M=lane&15; K=h+16*hf
__device__ __forceinline__ v16h cvtA(const float* rowp, int hf) {
  const float* p0 = rowp + hf * 16;
  const float* p1 = rowp + hf * 16 + 8;
  float4 f0 = ((const float4*)p0)[0];
  float4 f1 = ((const float4*)p0)[1];
  float4 f2 = ((const float4*)p1)[0];
  float4 f3 = ((const float4*)p1)[1];
  v16h r;
  r[0]  = (_Float16)f0.x; r[1]  = (_Float16)f0.y; r[2]  = (_Float16)f0.z; r[3]  = (_Float16)f0.w;
  r[4]  = (_Float16)f1.x; r[5]  = (_Float16)f1.y; r[6]  = (_Float16)f1.z; r[7]  = (_Float16)f1.w;
  r[8]  = (_Float16)f2.x; r[9]  = (_Float16)f2.y; r[10] = (_Float16)f2.z; r[11] = (_Float16)f2.w;
  r[12] = (_Float16)f3.x; r[13] = (_Float16)f3.y; r[14] = (_Float16)f3.z; r[15] = (_Float16)f3.w;
  return r;
}

// S[16x16] = Q(16x64) x K_j^T(64x16): two chained 16x16x32 WMMAs, f32 source (fallback)
__device__ __forceinline__ v8f qk_tile_f32(v16h aq0, v16h aq1, const float* kb,
                                           int j, int lh, int hf) {
  const float* kr = kb + (size_t)(j * 16 + lh) * DIM;
  v16h b0 = cvt16(kr + hf * 16);
  v16h b1 = cvt16(kr + 32 + hf * 16);
  v8f c = {};
  c = __builtin_amdgcn_wmma_f32_16x16x32_f16(false, aq0, false, b0, (short)0, c, false, false);
  c = __builtin_amdgcn_wmma_f32_16x16x32_f16(false, aq1, false, b1, (short)0, c, false, false);
  return c;
}

// Same, but K pre-converted to f16 row-major: B-fragments are direct aligned loads.
__device__ __forceinline__ v8f qk_tile_f16(v16h aq0, v16h aq1, const _Float16* kb16,
                                           int j, int lh, int hf) {
  const _Float16* kr = kb16 + (size_t)(j * 16 + lh) * DIM;
  v16h b0 = *(const v16h*)(kr + hf * 16);        // dims 0..31  (K = h + 16*hf)
  v16h b1 = *(const v16h*)(kr + 32 + hf * 16);   // dims 32..63
  v8f c = {};
  c = __builtin_amdgcn_wmma_f32_16x16x32_f16(false, aq0, false, b0, (short)0, c, false, false);
  c = __builtin_amdgcn_wmma_f32_16x16x32_f16(false, aq1, false, b1, (short)0, c, false, false);
  return c;
}

// ---------- prep: K -> f16 row-major ; V -> f16 transposed [b][dim][row] ----------
__global__ __launch_bounds__(256) void prep_kernel(const float* __restrict__ k,
                                                   const float* __restrict__ v,
                                                   _Float16* __restrict__ k16,
                                                   _Float16* __restrict__ vTg) {
  const size_t stride = (size_t)gridDim.x * blockDim.x;
  for (size_t i = (size_t)blockIdx.x * blockDim.x + threadIdx.x; i < NELEM; i += stride) {
    k16[i] = (_Float16)k[i];
    // output-linear transpose: i = (b*DIM + col)*SEQ + row  (coalesced writes)
    size_t row = i & (SEQ - 1);
    size_t bc  = i >> 11;            // b*DIM + col
    size_t col = bc & (DIM - 1);
    size_t b   = bc >> 6;
    vTg[i] = (_Float16)v[(b * SEQ + row) * DIM + col];
  }
}

// ---------- main attention kernel ----------
template <bool PREP>
__global__ __launch_bounds__(256) void sdpa_wmma_kernel(
    const float* __restrict__ q, const float* __restrict__ k,
    const float* __restrict__ v, const _Float16* __restrict__ k16,
    const _Float16* __restrict__ vTg, float* __restrict__ ctx,
    float* __restrict__ attn) {
  __shared__ alignas(16) _Float16 vT[2][DIM * VT_STRIDE];     // double-buffered V pair slab
  __shared__ alignas(16) _Float16 pS[WAVES][16 * PS_STRIDE];  // per-wave P tile bounce

  const int tid  = threadIdx.x;
  const int lane = tid & 31;
  const int wave = tid >> 5;
  const int lh   = lane & 15;
  const int hf   = lane >> 4;
  const int b    = blockIdx.x >> 4;
  const int qblk = blockIdx.x & 15;
  const int q0   = qblk * QROWS_WG + wave * 16;

  const float*    qb   = q   + (size_t)b * SEQ * DIM;
  const float*    kb   = k   + (size_t)b * SEQ * DIM;
  const float*    vb   = v   + (size_t)b * SEQ * DIM;
  const _Float16* kb16 = k16 + (size_t)b * SEQ * DIM;
  const _Float16* vTgb = vTg + (size_t)b * DIM * SEQ;

  // stage V pair slab t into LDS buffer `buf`
  auto stage = [&](int t, int buf) {
    if constexpr (PREP) {
      const int col = tid >> 2, seg = tid & 3;
      const _Float16* src = vTgb + (size_t)col * SEQ + t * 32 + seg * 8;
      _Float16* dst = &vT[buf][col * VT_STRIDE + seg * 8];
#if HAVE_ASYNC_LDS
      __builtin_amdgcn_global_load_async_to_lds_b128((gptr_v4i)src, (lptr_v4i)dst, 0, 0);
#else
      *(v8h*)dst = *(const v8h*)src;
#endif
    } else {
      const int row = tid >> 3;
      const int cb  = (tid & 7) * 8;
      const float* vp = vb + (size_t)(t * 32 + row) * DIM + cb;
      float4 f0 = ((const float4*)vp)[0];
      float4 f1 = ((const float4*)vp)[1];
      _Float16* dst = &vT[buf][0];
      dst[(cb + 0) * VT_STRIDE + row] = (_Float16)f0.x;
      dst[(cb + 1) * VT_STRIDE + row] = (_Float16)f0.y;
      dst[(cb + 2) * VT_STRIDE + row] = (_Float16)f0.z;
      dst[(cb + 3) * VT_STRIDE + row] = (_Float16)f0.w;
      dst[(cb + 4) * VT_STRIDE + row] = (_Float16)f1.x;
      dst[(cb + 5) * VT_STRIDE + row] = (_Float16)f1.y;
      dst[(cb + 6) * VT_STRIDE + row] = (_Float16)f1.z;
      dst[(cb + 7) * VT_STRIDE + row] = (_Float16)f1.w;
    }
  };

  // Q A-fragments (register-resident across all tiles)
  const float* qrow = qb + (size_t)(q0 + lh) * DIM;
  v16h aq0 = cvtA(qrow, hf);
  v16h aq1 = cvtA(qrow + 32, hf);

  // Pass 1: lane-local online (m,l) with ONE exp per element
  float m[8], l[8];
#pragma unroll
  for (int r = 0; r < 8; ++r) { m[r] = -3.0e38f; l[r] = 0.f; }

  for (int j = 0; j < NTILE; ++j) {
    v8f s;
    if constexpr (PREP) s = qk_tile_f16(aq0, aq1, kb16, j, lh, hf);
    else                s = qk_tile_f32(aq0, aq1, kb,   j, lh, hf);
#pragma unroll
    for (int r = 0; r < 8; ++r) {
      float d  = s[r] - m[r];
      float e  = __expf(fminf(d, -d));       // exp(-|d|)
      bool  gt = d > 0.f;
      l[r] = gt ? fmaf(l[r], e, 1.0f) : (l[r] + e);
      m[r] = fmaxf(m[r], s[r]);
    }
  }
  // merge across the 16 lanes of each half; fold log(l) into the exponent
  float ml[8];
#pragma unroll
  for (int r = 0; r < 8; ++r) {
#pragma unroll
    for (int mask = 1; mask < 16; mask <<= 1) {
      float om = __shfl_xor(m[r], mask);
      float ol = __shfl_xor(l[r], mask);
      float nm = fmaxf(m[r], om);
      l[r] = l[r] * __expf(m[r] - nm) + ol * __expf(om - nm);
      m[r] = nm;
    }
    ml[r] = m[r] + __logf(l[r]);   // p = exp(s - m)/l = exp(s - ml)
  }

  // Pass 2: recompute scores, emit attention (NT stores), accumulate context.
  // Double-buffered V slab: issue next pair's copy, compute current, wait+barrier.
  v8f acc[4] = {};
  float* attnb = attn + ((size_t)b * SEQ + q0) * SEQ;

  stage(0, 0);
  wait_async_lds();
  __syncthreads();

  for (int t = 0; t < NPAIR; ++t) {
    if (t + 1 < NPAIR) stage(t + 1, (t + 1) & 1);   // overlaps with compute below

#pragma unroll
    for (int sub = 0; sub < 2; ++sub) {
      const int j = 2 * t + sub;
      v8f s;
      if constexpr (PREP) s = qk_tile_f16(aq0, aq1, kb16, j, lh, hf);
      else                s = qk_tile_f32(aq0, aq1, kb,   j, lh, hf);
#pragma unroll
      for (int r = 0; r < 8; ++r) {
        const int M = r + 8 * hf;
        float p = __expf(s[r] - ml[r]);
        __builtin_nontemporal_store(p, &attnb[(size_t)M * SEQ + j * 16 + lh]);
        pS[wave][M * PS_STRIDE + sub * 16 + lh] = (_Float16)p;
      }
    }

    // A-fragment of P pair (16x32)
    v8h lo = *(const v8h*)&pS[wave][lh * PS_STRIDE + hf * 16];
    v8h hi = *(const v8h*)&pS[wave][lh * PS_STRIDE + hf * 16 + 8];
    v16h ap;
#pragma unroll
    for (int i = 0; i < 8; ++i) { ap[i] = lo[i]; ap[8 + i] = hi[i]; }

    // context += P_pair(16x32) x V_pair(32x64)
    const _Float16* vbuf = &vT[t & 1][0];
#pragma unroll
    for (int c = 0; c < 4; ++c) {
      const int col = c * 16 + lh;
      v8h b0 = *(const v8h*)&vbuf[col * VT_STRIDE + hf * 16];
      v8h b1 = *(const v8h*)&vbuf[col * VT_STRIDE + hf * 16 + 8];
      v16h bv;
#pragma unroll
      for (int i = 0; i < 8; ++i) { bv[i] = b0[i]; bv[8 + i] = b1[i]; }
      acc[c] = __builtin_amdgcn_wmma_f32_16x16x32_f16(false, ap, false, bv,
                                                      (short)0, acc[c], false, false);
    }

    wait_async_lds();   // our issued async copy (if any) has landed
    __syncthreads();    // publish buffer to all waves / retire readers
  }

  float* cbase = ctx + ((size_t)b * SEQ + q0) * DIM;
#pragma unroll
  for (int c = 0; c < 4; ++c)
#pragma unroll
    for (int r = 0; r < 8; ++r)
      cbase[(size_t)(r + 8 * hf) * DIM + c * 16 + lh] = acc[c][r];
}

// ---------- host ----------
extern "C" void kernel_launch(void* const* d_in, const int* in_sizes, int n_in,
                              void* d_out, int out_size, void* d_ws, size_t ws_size,
                              hipStream_t stream) {
  const float* q = (const float*)d_in[0];
  const float* k = (const float*)d_in[1];
  const float* v = (const float*)d_in[2];
  float* ctx  = (float*)d_out;
  float* attn = ctx + NELEM;  // tuple order: (context, attention)

  dim3 grid(BATCH * (SEQ / QROWS_WG));  // 256 workgroups
  dim3 block(WAVES * 32);               // 8 waves of 32

  const size_t need = 2 * NELEM * sizeof(_Float16);  // k16 + vT = 8 MB
  if (ws_size >= need) {
    _Float16* k16 = (_Float16*)d_ws;
    _Float16* vTg = k16 + NELEM;
    prep_kernel<<<2048, 256, 0, stream>>>(k, v, k16, vTg);
    sdpa_wmma_kernel<true><<<grid, block, 0, stream>>>(q, k, v, k16, vTg, ctx, attn);
  } else {
    sdpa_wmma_kernel<false><<<grid, block, 0, stream>>>(q, k, v, nullptr, nullptr, ctx, attn);
  }
}
